// SAKEFlowLayer_79955111182341
// MI455X (gfx1250) — matrix operand order repaired and checked by reading, not compiled, not run on hardware
//
#include <hip/hip_runtime.h>
#include <hip/hip_bf16.h>
#include <math.h>

// ---------------------------------------------------------------------------
// SAKE flow layer for gfx1250 (MI455X).
// Edge/node MLPs run on v_wmma_f32_16x16x32_f16 (wave32 fragments per CDNA5
// ISA layout tables). Weights are pre-packed into B-fragment-native layout.
// Working set (~33MB) is L2-resident (192MB); compute-bound on matrix cores.
// ---------------------------------------------------------------------------

typedef __attribute__((ext_vector_type(16))) _Float16 v16h;
typedef __attribute__((ext_vector_type(8)))  float    v8f;
using h16 = _Float16;

enum { Bd = 8, Nn = 128, NG = 129, NP = 144, HIDc = 64, NCOEF = 256, NHEAD = 4 };

__device__ __forceinline__ void ldsfence() {
  asm volatile("s_wait_dscnt 0x0" ::: "memory");
}

// A fragment (16x32 f16) from an LDS row-major tile. Per ISA 7.12.2:
// lane l holds row M=l%16; lanes 0-15 K-block offset 0, lanes 16-31 offset 8;
// VGPR pairs p<4 -> K base 0, p>=4 -> K base 16.
__device__ __forceinline__ v16h load_a(const h16* tile, int lda, int kt) {
  int lane = threadIdx.x & 31;
  int m = lane & 15, g = lane >> 4;
  const h16* row = tile + (size_t)m * lda + kt * 32 + g * 8;
  v16h a;
#pragma unroll
  for (int p = 0; p < 8; ++p) {
    int k = (p < 4 ? 0 : 16) + (p & 3) * 2;
    a[2 * p]     = row[k];
    a[2 * p + 1] = row[k + 1];
  }
  return a;
}

// B fragment from pre-packed weights: frag = 32 lanes x 16 halves, contiguous.
__device__ __forceinline__ v16h load_b(const h16* packed, int frag) {
  const uint4* q = (const uint4*)(packed + (size_t)frag * 512 + (threadIdx.x & 31) * 16);
  union { uint4 u[2]; v16h v; } c;
  c.u[0] = q[0]; c.u[1] = q[1];
  return c.v;
}

__device__ __forceinline__ v8f wmma(v16h a, v16h b, v8f c) {
  return __builtin_amdgcn_wmma_f32_16x16x32_f16(false, a, false, b, (short)0, c, false, false);
}

__device__ __forceinline__ float siluf(float x) { return x / (1.f + __expf(-x)); }

// ---------------------------------------------------------------------------
// Weight packing: out[f*512 + lane*16 + e] = W[kt*32 + (lane/16)*8 + (e%8) + (e/8)*16][nt*16 + lane%16]
// ---------------------------------------------------------------------------
__global__ void pack_w(const float* __restrict__ W, h16* __restrict__ out,
                       int K, int N, int KT, int NT) {
  int tid = blockIdx.x * 256 + threadIdx.x;
  int total = KT * NT * 512;
  if (tid >= total) return;
  int f = tid >> 9, r = tid & 511;
  int lane = r >> 4, e = r & 15;
  int kt = f / NT, nt = f % NT;
  int k = kt * 32 + (lane >> 4) * 8 + (e & 7) + (e >> 3) * 16;
  int n = nt * 16 + (lane & 15);
  out[tid] = (k < K && n < N) ? (h16)W[(size_t)k * N + n] : (h16)0.f;
}

// ---------------------------------------------------------------------------
// Init: h0 = concat(h, |x|^2); ghost(128) and pad(>=129) rows zero.
// ---------------------------------------------------------------------------
__global__ __launch_bounds__(64)
void k_init_h(const float* __restrict__ h_in, const float* __restrict__ x,
              float* __restrict__ hbuf, h16* __restrict__ hh) {
  int b = blockIdx.x / NP, node = blockIdx.x % NP, c = threadIdx.x;
  float val = 0.f;
  if (node < Nn) {
    if (c < 63) val = h_in[((size_t)(b * Nn) + node) * 63 + c];
    else {
      const float* xp = x + ((size_t)(b * Nn) + node) * 3;
      val = xp[0]*xp[0] + xp[1]*xp[1] + xp[2]*xp[2];
    }
  }
  size_t off = ((size_t)(b * NP) + node) * 64 + c;
  hbuf[off] = val;
  hh[off]   = (h16)val;
}

// Geometry (x is constant across steps): norm[i,j], xu[i,j,3]; invalid -> 0.
__global__ void k_geom(const float* __restrict__ x, float* __restrict__ normb,
                       float* __restrict__ xub) {
  int idx = blockIdx.x * 256 + threadIdx.x;
  if (idx >= Bd * NP * NP) return;
  int j = idx % NP, rem = idx / NP;
  int i = rem % NP, b = rem / NP;
  float xi0=0,xi1=0,xi2=0, xj0=0,xj1=0,xj2=0;
  if (i < Nn) { const float* p = x + ((size_t)(b*Nn)+i)*3; xi0=p[0]; xi1=p[1]; xi2=p[2]; }
  if (j < Nn) { const float* p = x + ((size_t)(b*Nn)+j)*3; xj0=p[0]; xj1=p[1]; xj2=p[2]; }
  bool valid = (i < NG) && (j < NG);
  float d0 = xj0-xi0, d1 = xj1-xi1, d2 = xj2-xi2;
  float nn = valid ? sqrtf(fmaxf(d0*d0 + d1*d1 + d2*d2, 0.f) + 1e-5f) : 0.f;
  normb[idx] = nn;
  float inv = valid ? 1.f/(nn + 1e-5f) : 0.f;
  xub[(size_t)idx*3+0] = d0*inv;
  xub[(size_t)idx*3+1] = d1*inv;
  xub[(size_t)idx*3+2] = d2*inv;
}

// ---------------------------------------------------------------------------
// Edge kernel: block = (b, i<128), 4 waves sweep 9 j-tiles of 16 edges.
// Per tile: GEMM1(+rbf filter) -> GEMM2(silu) -> GEMM3 = h_e -> store (f16)
//           + semantic logits (celu) + x_mixing(tanh) comb reduction.
// ---------------------------------------------------------------------------
__global__ __launch_bounds__(128)
void k_edge(const h16* __restrict__ hh, const float* __restrict__ normb,
            const float* __restrict__ xub,
            const h16* __restrict__ pW1, const float* __restrict__ b1,
            const h16* __restrict__ pW2, const float* __restrict__ b2,
            const h16* __restrict__ pW3, const float* __restrict__ b3,
            const h16* __restrict__ pWatt, const float* __restrict__ batt,
            const h16* __restrict__ pWmix,
            h16* __restrict__ heg, float* __restrict__ semb,
            float* __restrict__ combn) {
  int b = blockIdx.x >> 7, i = blockIdx.x & 127;
  int wave = threadIdx.x >> 5, lane = threadIdx.x & 31;
  int cl = lane & 15, g = lane >> 4;

  __shared__ __align__(16) h16 At[4][16][192];
  __shared__ __align__(16) h16 He[4][16][64];
  __shared__ float NormT[4][16];
  __shared__ float XuT[4][16][3];
  __shared__ float combW[4][768];   // per-wave private partials (deterministic)

  for (int t = lane; t < 768; t += 32) combW[wave][t] = 0.f;

  const float RSTART = 0.0067379470f;
  const float MSTEP  = (1.0f - RSTART) / 49.0f;
  const float RBETA  = 1.0f / ((0.04f*(1.0f-RSTART)) * (0.04f*(1.0f-RSTART)));

  const size_t rowIJ = ((size_t)(b * NP) + i) * NP;

  for (int jt = wave; jt < 9; jt += 4) {
    int j0 = jt * 16;
    if (lane < 16) {
      NormT[wave][lane] = normb[rowIJ + j0 + lane];
#pragma unroll
      for (int d = 0; d < 3; ++d)
        XuT[wave][lane][d] = xub[(rowIJ + j0 + lane) * 3 + d];
    }
    // A cols 0..127 = [h_j | h_i]
    for (int t = lane; t < 256; t += 32) {
      int row = t >> 4, cg = t & 15;
      const uint4* src = (cg < 8)
        ? (const uint4*)(hh + ((size_t)(b*NP) + j0 + row) * 64 + cg * 8)
        : (const uint4*)(hh + ((size_t)(b*NP) + i) * 64 + (cg - 8) * 8);
      *(uint4*)(&At[wave][row][cg * 8]) = *src;
    }
    ldsfence();

    // GEMM1: [16x128] x [128x64(50 valid)]
    v8f d1[4] = {};
#pragma unroll
    for (int kt = 0; kt < 4; ++kt) {
      v16h a = load_a(&At[wave][0][0], 192, kt);
#pragma unroll
      for (int nt = 0; nt < 4; ++nt) d1[nt] = wmma(a, load_b(pW1, kt*4+nt), d1[nt]);
    }
    // filt = rbf(norm)*hk -> A cols 128..177 ; col 178 = norm ; 179..191 = 0
#pragma unroll
    for (int nt = 0; nt < 4; ++nt) {
      int colb = nt * 16 + cl;
      if (colb < 50) {
        float bias = b1[colb];
        float mean = RSTART + colb * MSTEP;
#pragma unroll
        for (int r = 0; r < 8; ++r) {
          int rowj = r + 8 * g;
          float nrm = NormT[wave][rowj];
          float hk  = d1[nt][r] + bias;
          float td  = __expf(-nrm) - mean;
          At[wave][rowj][128 + colb] = (h16)(__expf(-RBETA * td * td) * hk);
        }
      }
    }
    if (lane < 16) {
      At[wave][lane][178] = (h16)NormT[wave][lane];
#pragma unroll
      for (int c2 = 179; c2 < 192; ++c2) At[wave][lane][c2] = (h16)0.f;
    }
    ldsfence();

    // GEMM2: [16x192] x [192x64] + bias, silu -> He (stage)
    v8f d2[4] = {};
#pragma unroll
    for (int kt = 0; kt < 6; ++kt) {
      v16h a = load_a(&At[wave][0][0], 192, kt);
#pragma unroll
      for (int nt = 0; nt < 4; ++nt) d2[nt] = wmma(a, load_b(pW2, kt*4+nt), d2[nt]);
    }
#pragma unroll
    for (int nt = 0; nt < 4; ++nt) {
      int colb = nt * 16 + cl; float bias = b2[colb];
#pragma unroll
      for (int r = 0; r < 8; ++r) He[wave][r + 8*g][colb] = (h16)siluf(d2[nt][r] + bias);
    }
    ldsfence();

    // GEMM3: [16x64] x [64x64] + bias = h_e ; write back into He
    v16h s0 = load_a(&He[wave][0][0], 64, 0);
    v16h s1 = load_a(&He[wave][0][0], 64, 1);
    v8f d3[4] = {};
#pragma unroll
    for (int nt = 0; nt < 4; ++nt) {
      d3[nt] = wmma(s0, load_b(pW3, nt),     d3[nt]);
      d3[nt] = wmma(s1, load_b(pW3, 4 + nt), d3[nt]);
    }
#pragma unroll
    for (int nt = 0; nt < 4; ++nt) {
      int colb = nt * 16 + cl; float bias = b3[colb];
#pragma unroll
      for (int r = 0; r < 8; ++r) He[wave][r + 8*g][colb] = (h16)(d3[nt][r] + bias);
    }
    ldsfence();

    // store h_e tile -> global (f16)
    for (int t = lane; t < 128; t += 32) {
      int row = t >> 3, ch = t & 7;
      ((uint4*)(heg + (rowIJ + j0 + row) * 64))[ch] = ((const uint4*)(&He[wave][row][0]))[ch];
    }

    v16h h0 = load_a(&He[wave][0][0], 64, 0);
    v16h h1 = load_a(&He[wave][0][0], 64, 1);

    // semantic attention logits: celu(h_e @ Watt + b, alpha=2)
    {
      v8f ds = {};
      ds = wmma(h0, load_b(pWatt, 0), ds);
      ds = wmma(h1, load_b(pWatt, 1), ds);
      if (cl < 4) {
        float bias = batt[cl];
#pragma unroll
        for (int r = 0; r < 8; ++r) {
          float z = ds[r] + bias;
          float cel = z > 0.f ? z : 2.f * (__expf(0.5f * z) - 1.f);
          semb[(rowIJ + j0 + r + 8*g) * 4 + cl] = cel;
        }
      }
    }

    // x_mixing: coeff = tanh(h_e @ Wmix); comb partial = sum_j coeff*xu
#pragma unroll
    for (int nt = 0; nt < 16; ++nt) {
      v8f dm = {};
      dm = wmma(h0, load_b(pWmix, nt),      dm);
      dm = wmma(h1, load_b(pWmix, 16 + nt), dm);
      float p0 = 0.f, p1 = 0.f, p2 = 0.f;
#pragma unroll
      for (int r = 0; r < 8; ++r) {
        float c = tanhf(dm[r]);
        int rowj = r + 8 * g;
        p0 += c * XuT[wave][rowj][0];
        p1 += c * XuT[wave][rowj][1];
        p2 += c * XuT[wave][rowj][2];
      }
      // merge the two half-wave groups (same output column c)
      p0 += __shfl_xor(p0, 16); p1 += __shfl_xor(p1, 16); p2 += __shfl_xor(p2, 16);
      if (g == 0) {
        int c = nt * 16 + cl;
        combW[wave][c*3+0] += p0;
        combW[wave][c*3+1] += p1;
        combW[wave][c*3+2] += p2;
      }
    }
  }
  __syncthreads();

  const float inv_n2 = 1.f / (129.f * 129.f);
  for (int c = threadIdx.x; c < 256; c += 128) {
    float a0 = 0.f, a1 = 0.f, a2 = 0.f;
#pragma unroll
    for (int w = 0; w < 4; ++w) {
      a0 += combW[w][c*3+0]; a1 += combW[w][c*3+1]; a2 += combW[w][c*3+2];
    }
    combn[((size_t)(b*NP) + i) * 256 + c] = (a0*a0 + a1*a1 + a2*a2) * inv_n2;
  }
}

// ---------------------------------------------------------------------------
// Attention: att = softmax_j(euc_logit + sem_logit) over j in [0,129).
// (exactly equal to normalized euc*sem product of the reference)
// ---------------------------------------------------------------------------
__global__ __launch_bounds__(128)
void k_att(const float* __restrict__ normb, const float* __restrict__ semb,
           const float* __restrict__ lgam, float* __restrict__ attb) {
  int b = blockIdx.x >> 7, i = blockIdx.x & 127;
  int t = threadIdx.x;
  __shared__ float red[128];
  __shared__ float gam[4];
  if (t < 4) gam[t] = __expf(lgam[t]);
  __syncthreads();
  size_t base = ((size_t)(b * NP) + i) * NP;
  float n0 = normb[base + t];
  float e0 = (t == i) ? 1e5f : 0.f;
  float n1 = (t == 0) ? normb[base + 128] : 0.f;
  float sl0[4], sl1[4];
#pragma unroll
  for (int h = 0; h < 4; ++h) sl0[h] = semb[(base + t) * 4 + h];
#pragma unroll
  for (int h = 0; h < 4; ++h) sl1[h] = (t == 0) ? semb[(base + 128) * 4 + h] : 0.f;

#pragma unroll
  for (int h = 0; h < 4; ++h) {
    float L0 = -(n0 + e0) * gam[h] + sl0[h];
    float L1 = (t == 0) ? (-n1 * gam[h] + sl1[h]) : -1e30f;
    red[t] = fmaxf(L0, L1); __syncthreads();
    for (int s = 64; s > 0; s >>= 1) { if (t < s) red[t] = fmaxf(red[t], red[t+s]); __syncthreads(); }
    float mx = red[0]; __syncthreads();
    float x0 = __expf(L0 - mx);
    float x1 = (t == 0) ? __expf(L1 - mx) : 0.f;
    red[t] = x0 + x1; __syncthreads();
    for (int s = 64; s > 0; s >>= 1) { if (t < s) red[t] += red[t+s]; __syncthreads(); }
    float inv = 1.f / red[0]; __syncthreads();
    attb[(base + t) * 4 + h] = x0 * inv;
    if (t == 0) attb[(base + 128) * 4 + h] = x1 * inv;
  }
}

// Aggregation: h_e[i, hid*4+head] = sum_{j<129} he[i,j,hid] * att[i,j,head]
__global__ __launch_bounds__(64)
void k_agg(const h16* __restrict__ heg, const float* __restrict__ attb,
           float* __restrict__ heb) {
  int b = blockIdx.x >> 7, i = blockIdx.x & 127;
  int t = threadIdx.x;
  __shared__ float attS[129 * 4];
  size_t base = ((size_t)(b * NP) + i) * NP;
  for (int idx = t; idx < 516; idx += 64) attS[idx] = attb[base * 4 + idx];
  __syncthreads();
  float acc[4] = {0.f, 0.f, 0.f, 0.f};
  const h16* her = heg + base * 64 + t;
  for (int j = 0; j < 129; ++j) {
    float hv = (float)her[(size_t)j * 64];
#pragma unroll
    for (int h = 0; h < 4; ++h) acc[h] += hv * attS[j * 4 + h];
  }
#pragma unroll
  for (int h = 0; h < 4; ++h)
    heb[((size_t)(b * NP) + i) * 256 + t * 4 + h] = acc[h];
}

// ---------------------------------------------------------------------------
// Node kernel: post_norm chain + node_mlp chain + residual. Nodes < 128 only
// (ghost stays zero, matching drop-and-repad of the reference).
// ---------------------------------------------------------------------------
__global__ __launch_bounds__(32)
void k_node(float* __restrict__ hbuf, h16* __restrict__ hh,
            const float* __restrict__ combn, const float* __restrict__ heb,
            const h16* __restrict__ pWp1, const float* __restrict__ bp1,
            const h16* __restrict__ pWp2, const float* __restrict__ bp2,
            const h16* __restrict__ pWn1, const float* __restrict__ bn1,
            const h16* __restrict__ pWn2, const float* __restrict__ bn2) {
  int b = blockIdx.x >> 3, lt = blockIdx.x & 7;
  int node0 = lt * 16;
  int lane = threadIdx.x, cl = lane & 15, g = lane >> 4;
  __shared__ __align__(16) h16 NA[16][384];
  __shared__ __align__(16) h16 CT[16][256];
  __shared__ __align__(16) h16 ST[16][64];

  for (int t = lane; t < 512; t += 32) {
    int row = t >> 5, ch = t & 31;
    const float* src = combn + ((size_t)(b*NP) + node0 + row) * 256 + ch * 8;
#pragma unroll
    for (int e = 0; e < 8; ++e) CT[row][ch*8+e] = (h16)src[e];
  }
  for (int t = lane; t < 128; t += 32) {
    int row = t >> 3, ch = t & 7;
    const float* src = hbuf + ((size_t)(b*NP) + node0 + row) * 64 + ch * 8;
#pragma unroll
    for (int e = 0; e < 8; ++e) NA[row][ch*8+e] = (h16)src[e];
  }
  for (int t = lane; t < 512; t += 32) {
    int row = t >> 5, ch = t & 31;
    const float* src = heb + ((size_t)(b*NP) + node0 + row) * 256 + ch * 8;
#pragma unroll
    for (int e = 0; e < 8; ++e) NA[row][64 + ch*8+e] = (h16)src[e];
  }
  ldsfence();

  // post_norm1 (256->64) silu
  v8f dp[4] = {};
  for (int kt = 0; kt < 8; ++kt) {
    v16h a = load_a(&CT[0][0], 256, kt);
#pragma unroll
    for (int nt = 0; nt < 4; ++nt) dp[nt] = wmma(a, load_b(pWp1, kt*4+nt), dp[nt]);
  }
#pragma unroll
  for (int nt = 0; nt < 4; ++nt) {
    int colb = nt*16+cl; float bias = bp1[colb];
#pragma unroll
    for (int r = 0; r < 8; ++r) ST[r+8*g][colb] = (h16)siluf(dp[nt][r] + bias);
  }
  ldsfence();
  // post_norm2 (64->64) silu -> NA cols 320..383 (h_comb)
  {
    v16h a0 = load_a(&ST[0][0], 64, 0), a1 = load_a(&ST[0][0], 64, 1);
    v8f dq[4] = {};
#pragma unroll
    for (int nt = 0; nt < 4; ++nt) {
      dq[nt] = wmma(a0, load_b(pWp2, nt),     dq[nt]);
      dq[nt] = wmma(a1, load_b(pWp2, 4 + nt), dq[nt]);
    }
#pragma unroll
    for (int nt = 0; nt < 4; ++nt) {
      int colb = nt*16+cl; float bias = bp2[colb];
#pragma unroll
      for (int r = 0; r < 8; ++r) NA[r+8*g][320+colb] = (h16)siluf(dq[nt][r] + bias);
    }
  }
  ldsfence();

  // node_mlp1 (384->64) silu
  v8f dn[4] = {};
  for (int kt = 0; kt < 12; ++kt) {
    v16h a = load_a(&NA[0][0], 384, kt);
#pragma unroll
    for (int nt = 0; nt < 4; ++nt) dn[nt] = wmma(a, load_b(pWn1, kt*4+nt), dn[nt]);
  }
#pragma unroll
  for (int nt = 0; nt < 4; ++nt) {
    int colb = nt*16+cl; float bias = bn1[colb];
#pragma unroll
    for (int r = 0; r < 8; ++r) ST[r+8*g][colb] = (h16)siluf(dn[nt][r] + bias);
  }
  ldsfence();
  // node_mlp2 (64->64) silu + residual
  v16h a0 = load_a(&ST[0][0], 64, 0), a1 = load_a(&ST[0][0], 64, 1);
  v8f dd[4] = {};
#pragma unroll
  for (int nt = 0; nt < 4; ++nt) {
    dd[nt] = wmma(a0, load_b(pWn2, nt),     dd[nt]);
    dd[nt] = wmma(a1, load_b(pWn2, 4 + nt), dd[nt]);
  }
#pragma unroll
  for (int nt = 0; nt < 4; ++nt) {
    int colb = nt*16+cl; float bias = bn2[colb];
#pragma unroll
    for (int r = 0; r < 8; ++r) {
      int node = node0 + r + 8*g;
      size_t off = ((size_t)(b*NP) + node) * 64 + colb;
      float hn = hbuf[off] + siluf(dd[nt][r] + bias);
      hbuf[off] = hn;
      hh[off]   = (h16)hn;
    }
  }
}

// ---------------------------------------------------------------------------
// Final coord kernel: coeff_ij = tanh(w2 . silu(W1 [h_j|h_i] + b)),
// delta_v[i] = mean_j (x_j - x_i) coeff_ij   (no ghost here, j<128)
// ---------------------------------------------------------------------------
__global__ __launch_bounds__(32)
void k_coord(const h16* __restrict__ hh, const float* __restrict__ x,
             const h16* __restrict__ pWc1, const float* __restrict__ bc1,
             const float* __restrict__ wc2, float* __restrict__ deltab) {
  int b = blockIdx.x >> 7, i = blockIdx.x & 127;
  int lane = threadIdx.x, cl = lane & 15, g = lane >> 4;
  __shared__ __align__(16) h16 CA[16][128];
  __shared__ float part[16][16];
  __shared__ float redc[16][3];
  float w2v[4], bv[4];
#pragma unroll
  for (int nt = 0; nt < 4; ++nt) { int colb = nt*16+cl; w2v[nt] = wc2[colb]; bv[nt] = bc1[colb]; }
  float av0 = 0.f, av1 = 0.f, av2 = 0.f;
  const float* xi = x + ((size_t)b * Nn + i) * 3;

  for (int jt = 0; jt < 8; ++jt) {
    int j0 = jt * 16;
    for (int t = lane; t < 256; t += 32) {
      int row = t >> 4, cg = t & 15;
      const uint4* src = (cg < 8)
        ? (const uint4*)(hh + ((size_t)(b*NP) + j0 + row) * 64 + cg * 8)
        : (const uint4*)(hh + ((size_t)(b*NP) + i) * 64 + (cg - 8) * 8);
      *(uint4*)(&CA[row][cg * 8]) = *src;
    }
    ldsfence();
    v8f dc[4] = {};
#pragma unroll
    for (int kt = 0; kt < 4; ++kt) {
      v16h a = load_a(&CA[0][0], 128, kt);
#pragma unroll
      for (int nt = 0; nt < 4; ++nt) dc[nt] = wmma(a, load_b(pWc1, kt*4+nt), dc[nt]);
    }
#pragma unroll
    for (int r = 0; r < 8; ++r) {
      float s = 0.f;
#pragma unroll
      for (int nt = 0; nt < 4; ++nt) s += siluf(dc[nt][r] + bv[nt]) * w2v[nt];
      part[r + 8*g][cl] = s;
    }
    ldsfence();
    if (lane < 16) {
      float pc = 0.f;
#pragma unroll
      for (int c = 0; c < 16; ++c) pc += part[lane][c];
      float coeff = tanhf(pc);
      int j = j0 + lane;
      const float* xj = x + ((size_t)b * Nn + j) * 3;
      av0 += (xj[0]-xi[0]) * coeff;
      av1 += (xj[1]-xi[1]) * coeff;
      av2 += (xj[2]-xi[2]) * coeff;
    }
    ldsfence();
  }
  if (lane < 16) { redc[lane][0]=av0; redc[lane][1]=av1; redc[lane][2]=av2; }
  ldsfence();
  if (lane == 0) {
    float d0=0.f,d1=0.f,d2=0.f;
#pragma unroll
    for (int l = 0; l < 16; ++l) { d0+=redc[l][0]; d1+=redc[l][1]; d2+=redc[l][2]; }
    float* dv = deltab + ((size_t)b * Nn + i) * 3;
    dv[0]=d0/128.f; dv[1]=d1/128.f; dv[2]=d2/128.f;
  }
}

// Velocity scaling + outputs: x copy, v_out, log_det.
__global__ __launch_bounds__(128)
void k_vel(const float* __restrict__ hbuf, const float* __restrict__ x,
           const float* __restrict__ v, const float* __restrict__ Wv1,
           const float* __restrict__ bv1, const float* __restrict__ wv2,
           const float* __restrict__ lamb, const float* __restrict__ deltab,
           float* __restrict__ out) {
  int b = blockIdx.x, t = threadIdx.x;
  __shared__ float red[128];
  const float* hrow = hbuf + ((size_t)(b * NP) + t) * 64;
  float hl[64];
#pragma unroll
  for (int k = 0; k < 64; ++k) hl[k] = hrow[k];
  float acc = 0.f;
  for (int hid = 0; hid < 64; ++hid) {
    float s = bv1[hid];
#pragma unroll 8
    for (int k = 0; k < 64; ++k) s += hl[k] * Wv1[k * 64 + hid];
    acc += siluf(s) * wv2[hid];
  }
  float m = tanhf(acc);
  float em = __expf(m);
  float lam = lamb[0];
#pragma unroll
  for (int d = 0; d < 3; ++d) {
    size_t idx = ((size_t)b * Nn + t) * 3 + d;
    float xv = x[idx], vv = v[idx];
    out[idx]        = xv;                               // x (unchanged)
    out[3072 + idx] = deltab[idx] + em * vv + lam * xv; // v_out
  }
  red[t] = m; __syncthreads();
  for (int s = 64; s > 0; s >>= 1) { if (t < s) red[t] += red[t + s]; __syncthreads(); }
  if (t == 0) out[6144 + b] = red[0] * 3.f;             // log_det = 3 * sum(m)
}

// ---------------------------------------------------------------------------
// Host launcher
// ---------------------------------------------------------------------------
extern "C" void kernel_launch(void* const* d_in, const int* in_sizes, int n_in,
                              void* d_out, int out_size, void* d_ws, size_t ws_size,
                              hipStream_t stream) {
  const float* h_in = (const float*)d_in[0];
  const float* x    = (const float*)d_in[1];
  const float* v    = (const float*)d_in[2];
  const float* W1   = (const float*)d_in[3];  const float* b1   = (const float*)d_in[4];
  const float* W2   = (const float*)d_in[5];  const float* b2   = (const float*)d_in[6];
  const float* W3   = (const float*)d_in[7];  const float* b3   = (const float*)d_in[8];
  const float* Watt = (const float*)d_in[9];  const float* batt = (const float*)d_in[10];
  const float* lgam = (const float*)d_in[11];
  const float* Wmix = (const float*)d_in[12];
  const float* Wp1  = (const float*)d_in[13]; const float* bp1  = (const float*)d_in[14];
  const float* Wp2  = (const float*)d_in[15]; const float* bp2  = (const float*)d_in[16];
  const float* Wn1  = (const float*)d_in[17]; const float* bn1  = (const float*)d_in[18];
  const float* Wn2  = (const float*)d_in[19]; const float* bn2  = (const float*)d_in[20];
  const float* Wc1  = (const float*)d_in[21]; const float* bc1  = (const float*)d_in[22];
  const float* wc2  = (const float*)d_in[23];
  const float* Wv1  = (const float*)d_in[24]; const float* bv1  = (const float*)d_in[25];
  const float* wv2  = (const float*)d_in[26]; const float* lamb = (const float*)d_in[27];

  char* ws = (char*)d_ws;
  size_t off = 0;
  auto alloc = [&](size_t bytes) -> char* {
    char* p = ws + off;
    off = (off + bytes + 255) & ~(size_t)255;
    return p;
  };
  float* hbuf  = (float*)alloc((size_t)Bd*NP*64*4);
  h16*   hh    = (h16*)  alloc((size_t)Bd*NP*64*2);
  float* normb = (float*)alloc((size_t)Bd*NP*NP*4);
  float* xub   = (float*)alloc((size_t)Bd*NP*NP*3*4);
  h16*   heg   = (h16*)  alloc((size_t)Bd*NP*NP*64*2);
  float* semb  = (float*)alloc((size_t)Bd*NP*NP*4*4);
  float* attb  = (float*)alloc((size_t)Bd*NP*NP*4*4);
  float* combn = (float*)alloc((size_t)Bd*NP*256*4);
  float* heb   = (float*)alloc((size_t)Bd*NP*256*4);
  float* deltab= (float*)alloc((size_t)Bd*Nn*3*4);
  h16* pW1   = (h16*)alloc(16*1024);
  h16* pW2   = (h16*)alloc(24*1024);
  h16* pW3   = (h16*)alloc(8*1024);
  h16* pWmix = (h16*)alloc(32*1024);
  h16* pWatt = (h16*)alloc(2*1024);
  h16* pWp1  = (h16*)alloc(32*1024);
  h16* pWp2  = (h16*)alloc(8*1024);
  h16* pWn1  = (h16*)alloc(48*1024);
  h16* pWn2  = (h16*)alloc(8*1024);
  h16* pWc1  = (h16*)alloc(16*1024);

  auto pack = [&](const float* W, h16* outp, int K, int N, int KT, int NT) {
    int total = KT * NT * 512;
    pack_w<<<(total + 255) / 256, 256, 0, stream>>>(W, outp, K, N, KT, NT);
  };
  pack(W1,   pW1,   128,  50,  4,  4);
  pack(W2,   pW2,   179,  64,  6,  4);
  pack(W3,   pW3,    64,  64,  2,  4);
  pack(Wmix, pWmix,  64, 256,  2, 16);
  pack(Watt, pWatt,  64,   4,  2,  1);
  pack(Wp1,  pWp1,  256,  64,  8,  4);
  pack(Wp2,  pWp2,   64,  64,  2,  4);
  pack(Wn1,  pWn1,  384,  64, 12,  4);
  pack(Wn2,  pWn2,   64,  64,  2,  4);
  pack(Wc1,  pWc1,  128,  64,  4,  4);

  k_init_h<<<Bd * NP, 64, 0, stream>>>(h_in, x, hbuf, hh);
  {
    int total = Bd * NP * NP;
    k_geom<<<(total + 255) / 256, 256, 0, stream>>>(x, normb, xub);
  }

  for (int step = 0; step < 4; ++step) {
    k_edge<<<Bd * Nn, 128, 0, stream>>>(hh, normb, xub,
                                        pW1, b1, pW2, b2, pW3, b3,
                                        pWatt, batt, pWmix,
                                        heg, semb, combn);
    k_att<<<Bd * Nn, 128, 0, stream>>>(normb, semb, lgam, attb);
    k_agg<<<Bd * Nn, 64, 0, stream>>>(heg, attb, heb);
    k_node<<<Bd * 8, 32, 0, stream>>>(hbuf, hh, combn, heb,
                                      pWp1, bp1, pWp2, bp2,
                                      pWn1, bn1, pWn2, bn2);
  }

  k_coord<<<Bd * Nn, 32, 0, stream>>>(hh, x, pWc1, bc1, wc2, deltab);
  k_vel<<<Bd, 128, 0, stream>>>(hbuf, x, v, Wv1, bv1, wv2, lamb, deltab,
                                (float*)d_out);
}
